// Decoder_57226144252590
// MI455X (gfx1250) — matrix operand order repaired and checked by reading, not compiled, not run on hardware
//
#include <hip/hip_runtime.h>
#include <hip/hip_bf16.h>

typedef __attribute__((ext_vector_type(16))) __bf16 v16bf;
typedef __attribute__((ext_vector_type(8)))  float  v8f;

#define B_   32
#define P_   196
#define T_   64
#define E_   512
#define H_   512
#define V_   32000
#define NG   2048     /* 4H */
#define KC_L0 48      /* K = 1536 = 2E + H */
#define KC_L1 32      /* K = 1024 = H + H  */
#define KC_V  16      /* K = 512           */
#define NT_V  2000    /* 32000/16 vocab N tiles */

// ---------------- helpers ----------------
__device__ __forceinline__ unsigned short f2bf(float f) {
  unsigned int u = __float_as_uint(f);
  u += 0x7FFFu + ((u >> 16) & 1u);          // round-to-nearest-even
  return (unsigned short)(u >> 16);
}
__device__ __forceinline__ float sigm(float x) { return 1.f / (1.f + __expf(-x)); }

__device__ __forceinline__ v8f wmma_bf16(v16bf a, v16bf b, v8f c) {
  return __builtin_amdgcn_wmma_f32_16x16x32_bf16(false, a, false, b, (short)0, c, false, false);
}

// A-fragment index for element A[row=b][col=k], 16-bit A 16x32, contiguous k order (k = 16*lane_half + j)
__device__ __forceinline__ int frag_idx(int b, int k, int KC) {
  int m  = b >> 4;
  int kc = k >> 5;
  int ko = k & 31;
  int lh = ko >> 4;                         // lane half
  int j  = ko & 15;                         // half index within v16bf
  int lane = (b & 15) + (lh << 4);
  return ((m * KC + kc) * 32 + lane) * 16 + j;
}

// ---------------- one-time kernels ----------------
__global__ __launch_bounds__(256) void mean_k(const float* __restrict__ enc, float* __restrict__ mean) {
  int b = blockIdx.x;
  for (int n = threadIdx.x; n < E_; n += 256) {
    float acc = 0.f;
    for (int p = 0; p < P_; ++p) acc += enc[((size_t)(b * P_ + p)) * E_ + n];
    mean[b * E_ + n] = acc * (1.f / (float)P_);
  }
}

__global__ __launch_bounds__(256) void initproj_k(const float* __restrict__ mean,
    const float* __restrict__ Wh, const float* __restrict__ bh,
    const float* __restrict__ Wc, const float* __restrict__ bc,
    float* __restrict__ hbuf, float* __restrict__ cbuf) {
  __shared__ float s_m[E_];
  int b = blockIdx.x;
  for (int k = threadIdx.x; k < E_; k += 256) s_m[k] = mean[b * E_ + k];
  __syncthreads();
  for (int k = threadIdx.x; k < 1024; k += 256) {
    float ah = bh[k], ac = bc[k];
    for (int e = 0; e < E_; ++e) { float m = s_m[e]; ah += m * Wh[e * 1024 + k]; ac += m * Wc[e * 1024 + k]; }
    // flat reshape [B, H*L] -> [L, B, H]
    int f = b * 1024 + k;
    int l = f >> 14, rem = f & 16383, bp = rem >> 9, jj = rem & 511;
    hbuf[(l * B_ + bp) * H_ + jj] = ah;
    cbuf[(l * B_ + bp) * H_ + jj] = ac;
  }
}

__global__ __launch_bounds__(256) void encatt_k(const float* __restrict__ enc,
    const float* __restrict__ Wenc, const float* __restrict__ benc, float* __restrict__ encA) {
  __shared__ float s_x[E_];
  size_t row = blockIdx.x;                  // b*P + p
  for (int k = threadIdx.x; k < E_; k += 256) s_x[k] = enc[row * E_ + k];
  __syncthreads();
  for (int n = threadIdx.x; n < E_; n += 256) {
    float acc = benc[n];
    for (int k = 0; k < E_; ++k) acc += s_x[k] * Wenc[k * E_ + n];
    encA[row * E_ + n] = acc;
  }
}

__global__ void biassum_k(const float* __restrict__ a, const float* __restrict__ b,
                          float* __restrict__ o, int n) {
  int i = blockIdx.x * blockDim.x + threadIdx.x;
  if (i < n) o[i] = a[i] + b[i];
}

// pack emb_W^T into B fragments (32x16 K-chunks, lanes 0-15: K=0..15, lanes 16-31: K=16..31)
__global__ void pack_vocab_k(const float* __restrict__ embW, unsigned short* __restrict__ Bv) {
  size_t idx = (size_t)blockIdx.x * blockDim.x + threadIdx.x;
  if (idx >= (size_t)NT_V * KC_V * 32 * 16) return;
  int j = idx & 15, lane = (idx >> 4) & 31, kc = (idx >> 9) & 15;
  int nt = (int)(idx >> 13);
  int v = nt * 16 + (lane & 15);
  int k = kc * 32 + ((lane >> 4) << 4) + j;
  Bv[idx] = f2bf(embW[(size_t)v * E_ + k]);
}

__global__ void pack_l0_k(const float* __restrict__ Wih, const float* __restrict__ Whh,
                          unsigned short* __restrict__ Bp) {
  size_t idx = (size_t)blockIdx.x * blockDim.x + threadIdx.x;
  if (idx >= (size_t)128 * KC_L0 * 32 * 16) return;
  int nt = (int)(idx / (KC_L0 * 512));
  int rem = (int)(idx % (KC_L0 * 512));
  int kc = rem >> 9, lane = (rem >> 4) & 31, j = rem & 15;
  int n = nt * 16 + (lane & 15);
  int k = kc * 32 + ((lane >> 4) << 4) + j;
  float s = (k < 1024) ? Wih[(size_t)n * 1024 + k] : Whh[(size_t)n * 512 + (k - 1024)];
  Bp[idx] = f2bf(s);
}

__global__ void pack_l1_k(const float* __restrict__ Wih, const float* __restrict__ Whh,
                          unsigned short* __restrict__ Bp) {
  size_t idx = (size_t)blockIdx.x * blockDim.x + threadIdx.x;
  if (idx >= (size_t)128 * KC_L1 * 32 * 16) return;
  int nt = (int)(idx / (KC_L1 * 512));
  int rem = (int)(idx % (KC_L1 * 512));
  int kc = rem >> 9, lane = (rem >> 4) & 31, j = rem & 15;
  int n = nt * 16 + (lane & 15);
  int k = kc * 32 + ((lane >> 4) << 4) + j;
  float s = (k < 512) ? Wih[(size_t)n * 512 + k] : Whh[(size_t)n * 512 + (k - 512)];
  Bp[idx] = f2bf(s);
}

// ---------------- per-step kernels ----------------
// attention + softmax + context + build A fragments for LSTM layer 0
__global__ __launch_bounds__(256) void attn_k(const float* __restrict__ encA,
    const float* __restrict__ enc, const int* __restrict__ caps,
    const float* __restrict__ embW, const float* __restrict__ Wdec,
    const float* __restrict__ bdec, const float* __restrict__ Wfull,
    const float* __restrict__ bfull, const float* __restrict__ hbuf,
    unsigned short* __restrict__ aF0, int t) {
  __shared__ float s_h1[E_], s_att2[E_], s_emb[E_], s_ctx[E_];
  __shared__ float s_alpha[P_ + 4];
  __shared__ float s_inv;
  int b = blockIdx.x, tid = threadIdx.x;
  int cap = caps[b * T_ + t];
  for (int k = tid; k < E_; k += 256) {
    s_h1[k]  = hbuf[(32 + b) * E_ + k];                 // layer-1 hidden
    s_emb[k] = embW[(size_t)cap * E_ + k];
  }
  __syncthreads();
  for (int n = tid; n < E_; n += 256) {
    float acc = bdec[n];
    for (int k = 0; k < E_; ++k) acc += s_h1[k] * Wdec[k * E_ + n];
    s_att2[n] = acc;
  }
  __syncthreads();
  int lane = tid & 31, wave = tid >> 5;
  for (int p = wave; p < P_; p += 8) {
    const float* row = encA + ((size_t)(b * P_ + p)) * E_;
    float acc = 0.f;
    for (int k = lane; k < E_; k += 32) {
      float x = row[k] + s_att2[k];
      acc += (x > 0.f ? x : 0.f) * Wfull[k];
    }
    for (int o = 16; o > 0; o >>= 1) acc += __shfl_xor(acc, o, 32);
    if (lane == 0) s_alpha[p] = acc + bfull[0];
  }
  __syncthreads();
  if (tid == 0) {
    float mx = -1e30f;
    for (int p = 0; p < P_; ++p) mx = fmaxf(mx, s_alpha[p]);
    float sm = 0.f;
    for (int p = 0; p < P_; ++p) { float e = __expf(s_alpha[p] - mx); s_alpha[p] = e; sm += e; }
    s_inv = 1.f / sm;
  }
  __syncthreads();
  for (int n = tid; n < E_; n += 256) {
    float acc = 0.f;
    const float* base = enc + ((size_t)b * P_) * E_ + n;
    for (int p = 0; p < P_; ++p) acc += s_alpha[p] * base[(size_t)p * E_];
    s_ctx[n] = acc * s_inv;
  }
  __syncthreads();
  // fill this batch-row's slice of the layer0 A fragments: A = [emb | context | h0_prev]
  int m = b >> 4, lb = b & 15;
  for (int idx = tid; idx < KC_L0 * 32; idx += 256) {
    int kc = idx >> 5, r = idx & 31, lh = r >> 4, j = r & 15;
    int ko = (lh << 4) + j;
    int k = kc * 32 + ko;
    float v;
    if (k < 512)       v = s_emb[k];
    else if (k < 1024) v = s_ctx[k - 512];
    else               v = hbuf[b * E_ + (k - 1024)];  // layer-0 hidden (prev)
    aF0[((m * KC_L0 + kc) * 32 + (lb + (lh << 4))) * 16 + j] = f2bf(v);
  }
}

// gate GEMM: [32 x K] (A frags) x [K x 2048] (packed B) -> gates fp32, + bias
__global__ __launch_bounds__(128) void gemm_gates_k(const unsigned short* __restrict__ aFrag,
    const unsigned short* __restrict__ Bp, const float* __restrict__ bias,
    float* __restrict__ gates, int KC) {
  int lane = threadIdx.x & 31;
  int wave = threadIdx.x >> 5;
  int nt = blockIdx.x * 4 + wave;                      // 0..127
  v8f c0 = {0.f, 0.f, 0.f, 0.f, 0.f, 0.f, 0.f, 0.f};
  v8f c1 = c0;
  const unsigned short* a0p = aFrag + (size_t)lane * 16;
  const unsigned short* a1p = aFrag + ((size_t)(KC * 32) + lane) * 16;
  const unsigned short* bp  = Bp + ((size_t)nt * KC * 32 + lane) * 16;
  for (int kc = 0; kc < KC; ++kc) {
    v16bf a0 = *(const v16bf*)a0p;
    v16bf a1 = *(const v16bf*)a1p;
    v16bf bb = *(const v16bf*)bp;
    c0 = wmma_bf16(a0, bb, c0);
    c1 = wmma_bf16(a1, bb, c1);
    a0p += 512; a1p += 512; bp += 512;
  }
  int n = nt * 16 + (lane & 15);
  int rb = (lane >> 4) * 8;
  float bn = bias[n];
#pragma unroll
  for (int r = 0; r < 8; ++r) {
    gates[(r + rb) * NG + n]      = c0[r] + bn;
    gates[(16 + r + rb) * NG + n] = c1[r] + bn;
  }
}

__global__ void cell0_k(const float* __restrict__ g, float* __restrict__ hbuf,
                        float* __restrict__ cbuf, unsigned short* __restrict__ aF1) {
  int idx = blockIdx.x * blockDim.x + threadIdx.x;    // 16384
  if (idx >= B_ * H_) return;
  int b = idx >> 9, j = idx & 511;
  float gi = g[b * NG + j], gf = g[b * NG + 512 + j];
  float gg = g[b * NG + 1024 + j], go = g[b * NG + 1536 + j];
  float cp = cbuf[b * H_ + j];
  float cn = sigm(gf) * cp + sigm(gi) * tanhf(gg);
  float hn = sigm(go) * tanhf(cn);
  cbuf[b * H_ + j] = cn;
  hbuf[b * H_ + j] = hn;
  float h1p = hbuf[(32 + b) * H_ + j];                 // layer-1 h (not yet updated)
  aF1[frag_idx(b, j, KC_L1)]       = f2bf(hn);
  aF1[frag_idx(b, 512 + j, KC_L1)] = f2bf(h1p);
}

__global__ void cell1_k(const float* __restrict__ g, float* __restrict__ hbuf,
                        float* __restrict__ cbuf, unsigned short* __restrict__ aFV) {
  int idx = blockIdx.x * blockDim.x + threadIdx.x;
  if (idx >= B_ * H_) return;
  int b = idx >> 9, j = idx & 511;
  float gi = g[b * NG + j], gf = g[b * NG + 512 + j];
  float gg = g[b * NG + 1024 + j], go = g[b * NG + 1536 + j];
  float cp = cbuf[(32 + b) * H_ + j];
  float cn = sigm(gf) * cp + sigm(gi) * tanhf(gg);
  float hn = sigm(go) * tanhf(cn);
  cbuf[(32 + b) * H_ + j] = cn;
  hbuf[(32 + b) * H_ + j] = hn;
  aFV[frag_idx(b, j, KC_V)] = f2bf(hn);
}

// tied vocab projection: [32 x 512] x [512 x 32000] -> out[b, t, v]
__global__ __launch_bounds__(128) void vocab_k(const unsigned short* __restrict__ aF,
    const unsigned short* __restrict__ Bv, const float* __restrict__ fcb,
    float* __restrict__ out, int t) {
  int lane = threadIdx.x & 31;
  int wave = threadIdx.x >> 5;
  int nt = blockIdx.x * 4 + wave;                      // 0..1999
  v8f c0 = {0.f, 0.f, 0.f, 0.f, 0.f, 0.f, 0.f, 0.f};
  v8f c1 = c0;
  const unsigned short* bp = Bv + ((size_t)nt * KC_V * 32 + lane) * 16;
#pragma unroll
  for (int kc = 0; kc < KC_V; ++kc) {
    __builtin_prefetch(bp + 512, 0, 1);                // next B fragment -> global_prefetch_b8
    v16bf a0 = *(const v16bf*)(aF + ((size_t)(kc * 32) + lane) * 16);
    v16bf a1 = *(const v16bf*)(aF + ((size_t)((KC_V + kc) * 32) + lane) * 16);
    v16bf bb = *(const v16bf*)bp;
    c0 = wmma_bf16(a0, bb, c0);
    c1 = wmma_bf16(a1, bb, c1);
    bp += 512;
  }
  int v = nt * 16 + (lane & 15);
  int rb = (lane >> 4) * 8;
  float bn = fcb[v];
#pragma unroll
  for (int r = 0; r < 8; ++r) {
    int b0 = r + rb, b1 = 16 + r + rb;
    out[((size_t)b0 * T_ + t) * V_ + v] = c0[r] + bn;
    out[((size_t)b1 * T_ + t) * V_ + v] = c1[r] + bn;
  }
}

// ---------------- workspace layout (bytes) ----------------
constexpr size_t OFF_ENCATT = 0;                         // 32*196*512*4   = 12,845,056
constexpr size_t OFF_BV  = OFF_ENCATT + 12845056;        // 2000*16*512*2  = 32,768,000
constexpr size_t OFF_B0  = OFF_BV + 32768000;            // 128*48*512*2   = 6,291,456
constexpr size_t OFF_B1  = OFF_B0 + 6291456;             // 128*32*512*2   = 4,194,304
constexpr size_t OFF_H   = OFF_B1 + 4194304;             // 2*32*512*4     = 131,072
constexpr size_t OFF_C   = OFF_H + 131072;               // 131,072
constexpr size_t OFF_G0  = OFF_C + 131072;               // 32*2048*4      = 262,144
constexpr size_t OFF_G1  = OFF_G0 + 262144;              // 262,144
constexpr size_t OFF_AF0 = OFF_G1 + 262144;              // 2*48*512*2     = 98,304
constexpr size_t OFF_AF1 = OFF_AF0 + 98304;              // 2*32*512*2     = 65,536
constexpr size_t OFF_AFV = OFF_AF1 + 65536;              // 2*16*512*2     = 32,768
constexpr size_t OFF_B0S = OFF_AFV + 32768;              // 8,192
constexpr size_t OFF_B1S = OFF_B0S + 8192;               // 8,192
constexpr size_t OFF_MEAN = OFF_B1S + 8192;              // 65,536

extern "C" void kernel_launch(void* const* d_in, const int* in_sizes, int n_in,
                              void* d_out, int out_size, void* d_ws, size_t ws_size,
                              hipStream_t stream) {
  const float* enc   = (const float*)d_in[0];
  const int*   caps  = (const int*)d_in[1];
  const float* embW  = (const float*)d_in[2];
  const float* fcb   = (const float*)d_in[3];
  const float* Wenc  = (const float*)d_in[4];
  const float* benc  = (const float*)d_in[5];
  const float* Wdec  = (const float*)d_in[6];
  const float* bdec  = (const float*)d_in[7];
  const float* Wfull = (const float*)d_in[8];
  const float* bfull = (const float*)d_in[9];
  const float* Wih0  = (const float*)d_in[10];
  const float* Whh0  = (const float*)d_in[11];
  const float* bih0  = (const float*)d_in[12];
  const float* bhh0  = (const float*)d_in[13];
  const float* Wih1  = (const float*)d_in[14];
  const float* Whh1  = (const float*)d_in[15];
  const float* bih1  = (const float*)d_in[16];
  const float* bhh1  = (const float*)d_in[17];
  const float* Winh  = (const float*)d_in[18];
  const float* binh  = (const float*)d_in[19];
  const float* Winc  = (const float*)d_in[20];
  const float* binc  = (const float*)d_in[21];

  char* ws = (char*)d_ws;
  float* encA = (float*)(ws + OFF_ENCATT);
  unsigned short* Bv  = (unsigned short*)(ws + OFF_BV);
  unsigned short* B0  = (unsigned short*)(ws + OFF_B0);
  unsigned short* B1  = (unsigned short*)(ws + OFF_B1);
  float* hbuf  = (float*)(ws + OFF_H);
  float* cbuf  = (float*)(ws + OFF_C);
  float* g0    = (float*)(ws + OFF_G0);
  float* g1    = (float*)(ws + OFF_G1);
  unsigned short* aF0 = (unsigned short*)(ws + OFF_AF0);
  unsigned short* aF1 = (unsigned short*)(ws + OFF_AF1);
  unsigned short* aFV = (unsigned short*)(ws + OFF_AFV);
  float* b0s   = (float*)(ws + OFF_B0S);
  float* b1s   = (float*)(ws + OFF_B1S);
  float* meanb = (float*)(ws + OFF_MEAN);
  float* out   = (float*)d_out;

  // one-time setup
  mean_k<<<B_, 256, 0, stream>>>(enc, meanb);
  initproj_k<<<B_, 256, 0, stream>>>(meanb, Winh, binh, Winc, binc, hbuf, cbuf);
  encatt_k<<<B_ * P_, 256, 0, stream>>>(enc, Wenc, benc, encA);
  biassum_k<<<8, 256, 0, stream>>>(bih0, bhh0, b0s, NG);
  biassum_k<<<8, 256, 0, stream>>>(bih1, bhh1, b1s, NG);
  pack_vocab_k<<<(NT_V * KC_V * 512 + 255) / 256, 256, 0, stream>>>(embW, Bv);
  pack_l0_k<<<(128 * KC_L0 * 512 + 255) / 256, 256, 0, stream>>>(Wih0, Whh0, B0);
  pack_l1_k<<<(128 * KC_L1 * 512 + 255) / 256, 256, 0, stream>>>(Wih1, Whh1, B1);

  // recurrent scan
  for (int t = 0; t < T_; ++t) {
    attn_k<<<B_, 256, 0, stream>>>(encA, enc, caps, embW, Wdec, bdec, Wfull, bfull, hbuf, aF0, t);
    gemm_gates_k<<<32, 128, 0, stream>>>(aF0, B0, b0s, g0, KC_L0);
    cell0_k<<<64, 256, 0, stream>>>(g0, hbuf, cbuf, aF1);
    gemm_gates_k<<<32, 128, 0, stream>>>(aF1, B1, b1s, g1, KC_L1);
    cell1_k<<<64, 256, 0, stream>>>(g1, hbuf, cbuf, aFV);
    vocab_k<<<NT_V / 4, 128, 0, stream>>>(aFV, Bv, fcb, out, t);
  }
  (void)in_sizes; (void)n_in; (void)out_size; (void)ws_size;
}
